// DACGLSTMExpert_5677946765421
// MI455X (gfx1250) — matrix operand order repaired and checked by reading, not compiled, not run on hardware
//
#include <hip/hip_runtime.h>
#include <hip/hip_bf16.h>

// CGLSTM recurrence on MI455X (gfx1250, wave32, WMMA).
// 18432 independent sequences, T=120, H=32, DIN=1.
// One wave = 16 sequences; recurrent matmuls via v_wmma_f32_16x16x32_f16
// (f16 operands, f32 accumulate); 8 WMMA / step / wave (i,f,g gates: 6 tiles
// of 16 gate columns, contextual gate: 2 tiles). Output gate of the LSTM is
// dead in the reference and is not computed.
// h state round-trips through a private per-wave LDS tile to convert the
// WMMA D layout (N across lanes) into the next step's A layout (M across
// lanes). Same-wave LDS ops are in-order -> no barriers in the time loop.

typedef _Float16 v16h __attribute__((ext_vector_type(16)));
typedef _Float16 v8h  __attribute__((ext_vector_type(8)));
typedef float    v8f  __attribute__((ext_vector_type(8)));
typedef float    v4f  __attribute__((ext_vector_type(4)));

#define T_STEPS 120
#define NV 9
#define NH 32
#define NB 2048
#define BVSEQ (NB*NV)          // 18432 sequences
#define WPB 4                  // waves per block
#define LH_STRIDE 40           // halves per h row (32 used + pad, 80B rows)

static __device__ __forceinline__ float sigm(float x) {
  // 1 / (1 + 2^(-x*log2 e))
  return __builtin_amdgcn_rcpf(1.0f + __builtin_amdgcn_exp2f(-1.44269504f * x));
}
static __device__ __forceinline__ float tanh_fast(float x) {
  // 1 - 2 / (1 + 2^(2x*log2 e))
  return 1.0f - 2.0f * __builtin_amdgcn_rcpf(1.0f + __builtin_amdgcn_exp2f(2.88539008f * x));
}

__global__ __launch_bounds__(WPB * 32)
void DACGLSTMExpert_kernel(const float* __restrict__ x,
                           const float* __restrict__ W_ih,
                           const float* __restrict__ W_hh,
                           const float* __restrict__ b_ih,
                           const float* __restrict__ b_hh,
                           const float* __restrict__ W_cg_x,
                           const float* __restrict__ W_cg_h,
                           const float* __restrict__ b_cg,
                           float* __restrict__ out) {
  __shared__ float    Lx[WPB][T_STEPS * 16];     // x tile, transposed [t][seq]
  __shared__ _Float16 Lh[WPB][16 * LH_STRIDE];   // h state, f16, [seq][hid]

  const int lane = threadIdx.x & 31;
  const int wb   = threadIdx.x >> 5;
  const int sb   = (blockIdx.x * WPB + wb) * 16; // sequence base of this wave
  if (sb >= BVSEQ) return;                       // never taken (exact tiling)
  const int col = lane & 15;                     // N position within a tile
  const int hi  = lane >> 4;                     // lane half

  // ---- stage x tile (16 seq x 120 t) into LDS, transposed ----
  for (int i = lane; i < 16 * T_STEPS; i += 32) {
    const int s = i / T_STEPS, t = i % T_STEPS;
    Lx[wb][t * 16 + s] = x[(size_t)(sb + s) * T_STEPS + t];
  }
  for (int i = lane; i < 16 * LH_STRIDE; i += 32) Lh[wb][i] = (_Float16)0.0f;

  // ---- per-lane input-projection scalars (DIN==1: matmul degenerates) ----
  // gate tiles n=0..5 cover gates [0,96) = i,f,g ; output gate is dead.
  float wih[6], bias[6];
#pragma unroll
  for (int n = 0; n < 6; ++n) {
    const int g = 16 * n + col;
    wih[n]  = W_ih[g];                 // W_ih is [128,1]
    bias[n] = b_ih[g] + b_hh[g];
  }
  float wcgx[2], bcgv[2];
#pragma unroll
  for (int t2 = 0; t2 < 2; ++t2) {
    wcgx[t2] = W_cg_x[16 * t2 + col];
    bcgv[t2] = b_cg[16 * t2 + col];
  }

  // ---- recurrent weights as resident f16 WMMA B operands ----
  // B layout (32x16, KxN): lanes 0-15 hold K=0..15 (N=col), lanes 16-31 hold
  // K=16..31; element j of v16h = K = 16*hi + j. B[k][n] = W[n][k].
  v16h Bg[6], Bc[2];
#pragma unroll
  for (int n = 0; n < 6; ++n) {
    const float* wr = W_hh + (size_t)(16 * n + col) * NH + 16 * hi;
#pragma unroll
    for (int j = 0; j < 16; ++j) Bg[n][j] = (_Float16)wr[j];
  }
#pragma unroll
  for (int t2 = 0; t2 < 2; ++t2) {
    const float* wr = W_cg_h + (size_t)(16 * t2 + col) * NH + 16 * hi;
#pragma unroll
    for (int j = 0; j < 16; ++j) Bc[t2][j] = (_Float16)wr[j];
  }

  // ---- state (D/C layout: element e -> seq = sb + e + 8*hi, hid = col+16*t2)
  v8f cst[2], hnew[2];
#pragma unroll
  for (int t2 = 0; t2 < 2; ++t2)
#pragma unroll
    for (int e = 0; e < 8; ++e) { cst[t2][e] = 0.0f; hnew[t2][e] = 0.0f; }

  // ================= time loop (serial recurrence) =================
#pragma unroll 1
  for (int t = 0; t < T_STEPS; ++t) {
    // x_t for the 8 sequences this lane's D rows cover (LDS broadcast)
    const float* xp = &Lx[wb][t * 16 + 8 * hi];
    const v4f xa = *(const v4f*)xp;
    const v4f xb = *(const v4f*)(xp + 4);
    float xs[8] = {xa.x, xa.y, xa.z, xa.w, xb.x, xb.y, xb.z, xb.w};

    // A operand = h_prev (16x32 f16). A layout: lane row M=col,
    // elems 0..15 -> K = 16*hi + j.
    const _Float16* hp = &Lh[wb][col * LH_STRIDE];
    const v8h a0 = *(const v8h*)(hp + 16 * hi);
    const v8h a1 = *(const v8h*)(hp + 16 * hi + 8);
    v16h A;
#pragma unroll
    for (int j = 0; j < 8; ++j) { A[j] = a0[j]; A[8 + j] = a1[j]; }

    // C init = x-projection + biases, then WMMA accumulate h @ W^T
    v8f acc[6], accg[2];
#pragma unroll
    for (int n = 0; n < 6; ++n) {
#pragma unroll
      for (int e = 0; e < 8; ++e) acc[n][e] = xs[e] * wih[n] + bias[n];
      acc[n] = __builtin_amdgcn_wmma_f32_16x16x32_f16(
          false, A, false, Bg[n], (short)0, acc[n], false, false);
    }
#pragma unroll
    for (int t2 = 0; t2 < 2; ++t2) {
#pragma unroll
      for (int e = 0; e < 8; ++e) accg[t2][e] = xs[e] * wcgx[t2] + bcgv[t2];
      accg[t2] = __builtin_amdgcn_wmma_f32_16x16x32_f16(
          false, A, false, Bc[t2], (short)0, accg[t2], false, false);
    }

    // elementwise gates + state update; write h_new (f16) back to LDS
#pragma unroll
    for (int t2 = 0; t2 < 2; ++t2) {
#pragma unroll
      for (int e = 0; e < 8; ++e) {
        const float ig = sigm(acc[0 + t2][e]);
        const float fg = sigm(acc[2 + t2][e]);
        const float gg = tanh_fast(acc[4 + t2][e]);
        const float cv = fg * cst[t2][e] + ig * gg;
        const float cg = sigm(accg[t2][e]);       // uses h_prev: same A
        const float hv = cg * tanh_fast(cv);
        cst[t2][e]  = cv;
        hnew[t2][e] = hv;
        Lh[wb][(e + 8 * hi) * LH_STRIDE + col + 16 * t2] = (_Float16)hv;
      }
    }
  }

  // ---- write h_final: out flat = [BV][H] == [B][V*H] ----
#pragma unroll
  for (int t2 = 0; t2 < 2; ++t2)
#pragma unroll
    for (int e = 0; e < 8; ++e)
      out[(size_t)(sb + e + 8 * hi) * NH + col + 16 * t2] = hnew[t2][e];
}

extern "C" void kernel_launch(void* const* d_in, const int* in_sizes, int n_in,
                              void* d_out, int out_size, void* d_ws, size_t ws_size,
                              hipStream_t stream) {
  (void)in_sizes; (void)n_in; (void)d_ws; (void)ws_size; (void)out_size;
  const float* x      = (const float*)d_in[0];
  const float* W_ih   = (const float*)d_in[1];
  const float* W_hh   = (const float*)d_in[2];
  const float* b_ih   = (const float*)d_in[3];
  const float* b_hh   = (const float*)d_in[4];
  const float* W_cg_x = (const float*)d_in[5];
  const float* W_cg_h = (const float*)d_in[6];
  const float* b_cg   = (const float*)d_in[7];
  float* out = (float*)d_out;

  const int waves  = BVSEQ / 16;          // 1152
  const int blocks = waves / WPB;         // 288
  DACGLSTMExpert_kernel<<<blocks, WPB * 32, 0, stream>>>(
      x, W_ih, W_hh, b_ih, b_hh, W_cg_x, W_cg_h, b_cg, out);
}
